// RTLayer_11106785427560
// MI455X (gfx1250) — matrix operand order repaired and checked by reading, not compiled, not run on hardware
//
#include <hip/hip_runtime.h>

// ---------------------------------------------------------------------------
// CDNA5 (gfx1250) relational-transformer layer.
// All heavy GEMMs use v_wmma_f32_16x16x32_bf16 (fp32 accumulate).
// Weights pre-converted to bf16 once -> B fragments are raw b128 loads.
// ---------------------------------------------------------------------------

#define BB 2
#define NN 384
#define ND 128     // NODE_DIM
#define NH 256     // NODE_HIDDEN
#define ED 64      // EDGE_DIM
#define EH1 384
#define EH2 128
#define HEADS 8
#define SCALEV 0.17677669529663687f   // 1/sqrt(32)
#define EPS 1e-5f

typedef __attribute__((ext_vector_type(16))) __bf16          v16bf;
typedef __attribute__((ext_vector_type(8)))  float           v8f;
typedef __attribute__((ext_vector_type(4)))  float           f4_t;
typedef __attribute__((ext_vector_type(8)))  unsigned short  us8_t;
typedef __attribute__((ext_vector_type(16))) unsigned short  us16_t;

#define GAS __attribute__((address_space(1)))

__device__ __forceinline__ const GAS f4_t* g4(const float* p) {
  return (const GAS f4_t*)(unsigned long long)p;
}
__device__ __forceinline__ const GAS float* gp(const float* p) {
  return (const GAS float*)(unsigned long long)p;
}
__device__ __forceinline__ GAS float* gpw(float* p) {
  return (GAS float*)(unsigned long long)p;
}
__device__ __forceinline__ float gload(const float* p) { return *gp(p); }
__device__ __forceinline__ const GAS us8_t* gu8(const unsigned short* p) {
  return (const GAS us8_t*)(unsigned long long)p;
}

// cheap bf16 conversion: round-half-up (add 0x8000, truncate)
__device__ __forceinline__ unsigned short bfbits(float f) {
  unsigned u = __builtin_bit_cast(unsigned, f) + 0x8000u;
  return (unsigned short)(u >> 16);
}
__device__ __forceinline__ unsigned pack2(float a, float b) {
  unsigned ua = __builtin_bit_cast(unsigned, a) + 0x8000u;
  unsigned ub = __builtin_bit_cast(unsigned, b) + 0x8000u;
  return (ua >> 16) | (ub & 0xffff0000u);
}
__device__ __forceinline__ v16bf us_to_bf(us16_t u) {
  union { us16_t u; v16bf v; } c; c.u = u; return c.v;
}

// A fragment (16x32 bf16) from LDS bf16 row-major (p = row + k0).
__device__ __forceinline__ v16bf afrag_lds(const unsigned short* p, int hi) {
  us8_t lo = *(const us8_t*)(p + hi * 16);
  us8_t hh = *(const us8_t*)(p + 8 + hi * 16);
  us16_t u = __builtin_shufflevector(lo, hh, 0,1,2,3,4,5,6,7,8,9,10,11,12,13,14,15);
  return us_to_bf(u);
}
// A fragment built directly from global fp32 (p = row + k0)
__device__ __forceinline__ v16bf afrag_g(const float* p, int hi) {
  const GAS f4_t* a = g4(p + hi * 16);
  const GAS f4_t* b = g4(p + 8 + hi * 16);
  f4_t f0 = a[0], f1 = a[1], f2 = b[0], f3 = b[1];
  union { unsigned u[8]; us16_t v; } c;
  c.u[0] = pack2(f0[0], f0[1]); c.u[1] = pack2(f0[2], f0[3]);
  c.u[2] = pack2(f1[0], f1[1]); c.u[3] = pack2(f1[2], f1[3]);
  c.u[4] = pack2(f2[0], f2[1]); c.u[5] = pack2(f2[2], f2[3]);
  c.u[6] = pack2(f3[0], f3[1]); c.u[7] = pack2(f3[2], f3[3]);
  return us_to_bf(c.v);
}
// B fragment (32x16 bf16) from pre-converted bf16 weights:
// p = Wb + n*K + k0 + hi*16  (16 consecutive bf16 = two b128 loads)
__device__ __forceinline__ v16bf bfrag_b(const unsigned short* p) {
  const GAS us8_t* q = gu8(p);
  us16_t u = __builtin_shufflevector(q[0], q[1], 0,1,2,3,4,5,6,7,8,9,10,11,12,13,14,15);
  return us_to_bf(u);
}
__device__ __forceinline__ v8f wmma_bf16(v16bf a, v16bf b, v8f c) {
  return __builtin_amdgcn_wmma_f32_16x16x32_bf16(false, a, false, b, (short)0, c,
                                                 false, false);
}

// ---------------------------------------------------------------------------
// Kernel 0: one-time fp32 -> bf16 weight conversion into workspace
// ---------------------------------------------------------------------------
struct CvtDesc {
  const float* src[10];
  unsigned cnt4[10];   // number of float4 groups per segment
  unsigned dsh[10];    // dst offset in ushorts (even, 32B aligned)
};

__global__ void cvt_weights_kernel(CvtDesc d, unsigned short* wbf) {
  const int tid = blockIdx.x * blockDim.x + threadIdx.x;
  const int stride = gridDim.x * blockDim.x;
  GAS unsigned* o = (GAS unsigned*)(unsigned long long)wbf;
#pragma unroll 1
  for (int s = 0; s < 10; ++s) {
    const GAS f4_t* src = g4(d.src[s]);
    const unsigned n = d.cnt4[s];
    const unsigned ubase = d.dsh[s] >> 1;
    for (unsigned t = tid; t < n; t += stride) {
      f4_t f = src[t];
      o[ubase + 2 * t + 0] = pack2(f[0], f[1]);
      o[ubase + 2 * t + 1] = pack2(f[2], f[3]);
    }
  }
}

// ---------------------------------------------------------------------------
// Kernel 1: node projections qn/kn/vn = x @ W^T  (768x128 @ 128x128, x3)
// ---------------------------------------------------------------------------
__device__ __forceinline__ void proj_one(const v16bf* a, const unsigned short* wb,
                                         float* out, int m0, int ln, int hi) {
#pragma unroll
  for (int nt = 0; nt < 8; ++nt) {
    v8f c;
#pragma unroll
    for (int r = 0; r < 8; ++r) c[r] = 0.f;
#pragma unroll
    for (int kc = 0; kc < 4; ++kc)
      c = wmma_bf16(a[kc], bfrag_b(wb + (nt * 16 + ln) * ND + kc * 32 + hi * 16), c);
#pragma unroll
    for (int r = 0; r < 8; ++r)
      gpw(out)[(m0 + r + 8 * hi) * ND + nt * 16 + ln] = c[r];
  }
}

__global__ void node_proj_kernel(const float* __restrict__ x,
                                 const unsigned short* __restrict__ wqb,
                                 const unsigned short* __restrict__ wkb,
                                 const unsigned short* __restrict__ wvb,
                                 float* __restrict__ qn,
                                 float* __restrict__ kn,
                                 float* __restrict__ vn) {
  const int m0 = blockIdx.x * 16;
  const int lane = threadIdx.x & 31, ln = lane & 15, hi = lane >> 4;
  v16bf a[4];
#pragma unroll
  for (int kc = 0; kc < 4; ++kc)
    a[kc] = afrag_g(x + (m0 + ln) * ND + kc * 32, hi);
  proj_one(a, wqb, qn, m0, ln, hi);
  proj_one(a, wkb, kn, m0, ln, hi);
  proj_one(a, wvb, vn, m0, ln, hi);
}

// ---------------------------------------------------------------------------
// Kernel 2: edge-augmented attention, fully fused per (b, i).
// grid (N, B) x 256 threads (8 waves). adj[b,i,:,:] staged once in LDS (bf16).
// ---------------------------------------------------------------------------
__global__ void attention_kernel(const float* __restrict__ adj,
                                 const float* __restrict__ qn,
                                 const float* __restrict__ kn,
                                 const float* __restrict__ vn,
                                 const unsigned short* __restrict__ weqb,
                                 const unsigned short* __restrict__ wekb,
                                 const unsigned short* __restrict__ wevb,
                                 float* __restrict__ attn_out) {
  const int i = blockIdx.x, bz = blockIdx.y;
  const int tid = threadIdx.x;
  const int w = tid >> 5, lane = tid & 31, ln = lane & 15, hi = lane >> 4;

  __shared__ __align__(16) unsigned short shAdj[NN * ED];  // 48 KB bf16
  __shared__ float redA[8], redB[8], shOut[16];

  // stage adj row-block -> LDS bf16
  {
    const GAS f4_t* src = g4(adj + ((size_t)(bz * NN + i) * NN) * ED);
    unsigned* dst = (unsigned*)shAdj;
    for (int it = tid; it < NN * ED / 4; it += 256) {
      f4_t f = src[it];
      dst[it * 2 + 0] = pack2(f[0], f[1]);
      dst[it * 2 + 1] = pack2(f[2], f[3]);
    }
  }
  __syncthreads();

  const int rowbase = (bz * NN + i) * ND;
  for (int h = 0; h < HEADS; ++h) {
    if (tid < 16) shOut[tid] = 0.f;
    __syncthreads();

    const int hc = h * 16 + ln;
    const float qb = gload(qn + rowbase + hc);

    v8f sreg[3], vful[3];
    float lmax = -1e30f;
#pragma unroll
    for (int tt = 0; tt < 3; ++tt) {
      const int j0 = (w * 3 + tt) * 16;
      const unsigned short* arow = shAdj + (j0 + ln) * ED;
      v8f cq, ck, cv;
#pragma unroll
      for (int r = 0; r < 8; ++r) { cq[r] = 0.f; ck[r] = 0.f; cv[r] = 0.f; }
#pragma unroll
      for (int kc = 0; kc < 2; ++kc) {
        v16bf a = afrag_lds(arow + kc * 32, hi);
        cq = wmma_bf16(a, bfrag_b(weqb + hc * ED + kc * 32 + hi * 16), cq);
        ck = wmma_bf16(a, bfrag_b(wekb + hc * ED + kc * 32 + hi * 16), ck);
        cv = wmma_bf16(a, bfrag_b(wevb + hc * ED + kc * 32 + hi * 16), cv);
      }
      v8f s;
#pragma unroll
      for (int r = 0; r < 8; ++r) {
        const int j = j0 + r + 8 * hi;
        const float kb = gload(kn + (bz * NN + j) * ND + hc);
        const float vb = gload(vn + (bz * NN + j) * ND + hc);
        s[r] = (cq[r] + qb) * (ck[r] + kb);
        vful[tt][r] = cv[r] + vb;
      }
      // reduce over the 16 channel lanes of each half-wave group
#pragma unroll
      for (int r = 0; r < 8; ++r) {
        float sv = s[r];
        sv += __shfl_xor(sv, 1, 32);
        sv += __shfl_xor(sv, 2, 32);
        sv += __shfl_xor(sv, 4, 32);
        sv += __shfl_xor(sv, 8, 32);
        sv *= SCALEV;
        sreg[tt][r] = sv;
        lmax = fmaxf(lmax, sv);
      }
    }
    lmax = fmaxf(lmax, __shfl_xor(lmax, 16, 32));
    if (lane == 0) redA[w] = lmax;
    __syncthreads();
    float mx = redA[0];
#pragma unroll
    for (int q = 1; q < 8; ++q) mx = fmaxf(mx, redA[q]);

    // cache exp(s - mx) in sreg
    float lsum = 0.f;
#pragma unroll
    for (int tt = 0; tt < 3; ++tt)
#pragma unroll
      for (int r = 0; r < 8; ++r) {
        const float e = __expf(sreg[tt][r] - mx);
        sreg[tt][r] = e;
        lsum += e;
      }
    lsum += __shfl_xor(lsum, 16, 32);
    if (lane == 0) redB[w] = lsum;
    __syncthreads();
    float den = 0.f;
#pragma unroll
    for (int q = 0; q < 8; ++q) den += redB[q];
    const float inv = 1.f / den;

    float po = 0.f;
#pragma unroll
    for (int tt = 0; tt < 3; ++tt)
#pragma unroll
      for (int r = 0; r < 8; ++r)
        po += sreg[tt][r] * vful[tt][r];
    po *= inv;
    po += __shfl_xor(po, 16, 32);
    if (lane < 16) atomicAdd(&shOut[ln], po);
    __syncthreads();
    // channel interleave: out[b,i, d*H + h]
    if (tid < 16) gpw(attn_out)[rowbase + tid * HEADS + h] = shOut[tid];
    __syncthreads();
  }
}

// ---------------------------------------------------------------------------
// Kernel 3: node post-attention: fc1 + LN + FFN + LN (tiny, VALU).
// Also emits bf16 copy of x_out for the edge kernel's staging.
// ---------------------------------------------------------------------------
__global__ void node_post_kernel(const float* __restrict__ x,
                                 const float* __restrict__ attn,
                                 const float* __restrict__ w1, const float* __restrict__ b1,
                                 const float* __restrict__ g1, const float* __restrict__ be1,
                                 const float* __restrict__ w2, const float* __restrict__ b2,
                                 const float* __restrict__ w3, const float* __restrict__ b3,
                                 const float* __restrict__ g2, const float* __restrict__ be2,
                                 float* __restrict__ xout,
                                 unsigned short* __restrict__ xoutb) {
  const int row = blockIdx.x;
  const int tid = threadIdx.x;
  __shared__ float sA[ND], sX[ND], sT[ND], sH[NH], red[2];
  if (tid < ND) {
    sA[tid] = gload(attn + row * ND + tid);
    sX[tid] = gload(x + row * ND + tid);
  }
  if (tid < 2) red[tid] = 0.f;
  __syncthreads();
  float t = 0.f;
  if (tid < ND) {
    const GAS float* w1g = gp(w1 + tid * ND);
    float r = gload(b1 + tid);
    for (int k = 0; k < ND; ++k) r += sA[k] * w1g[k];
    t = sX[tid] + r;
    atomicAdd(&red[0], t);
    atomicAdd(&red[1], t * t);
  }
  __syncthreads();
  float mean = red[0] * (1.f / ND);
  float var  = red[1] * (1.f / ND) - mean * mean;
  float rs = rsqrtf(var + EPS);
  if (tid < ND) sT[tid] = (t - mean) * rs * gload(g1 + tid) + gload(be1 + tid);
  __syncthreads();
  {
    const GAS float* w2g = gp(w2 + tid * ND);
    float hh = gload(b2 + tid);
    for (int c = 0; c < ND; ++c) hh += sT[c] * w2g[c];
    sH[tid] = fmaxf(hh, 0.f);
  }
  if (tid == 0) { red[0] = 0.f; red[1] = 0.f; }
  __syncthreads();
  float t2 = 0.f;
  if (tid < ND) {
    const GAS float* w3g = gp(w3 + tid * NH);
    float r = gload(b3 + tid);
    for (int j = 0; j < NH; ++j) r += sH[j] * w3g[j];
    t2 = sT[tid] + r;
    atomicAdd(&red[0], t2);
    atomicAdd(&red[1], t2 * t2);
  }
  __syncthreads();
  mean = red[0] * (1.f / ND);
  var  = red[1] * (1.f / ND) - mean * mean;
  rs = rsqrtf(var + EPS);
  if (tid < ND) {
    const float v = (t2 - mean) * rs * gload(g2 + tid) + gload(be2 + tid);
    gpw(xout)[row * ND + tid] = v;
    ((GAS unsigned short*)(unsigned long long)xoutb)[row * ND + tid] = bfbits(v);
  }
}

// ---------------------------------------------------------------------------
// Kernel 4: fused edge update. grid (6, 384, 2) x 128 threads (4 waves).
// 4 chained WMMA GEMMs + bias/ReLU/residual/LayerNorm, intermediates in LDS.
// ---------------------------------------------------------------------------
__global__ void edge_kernel(const float* __restrict__ adj,
                            const float* __restrict__ xf,
                            const unsigned short* __restrict__ xfb,
                            const float* __restrict__ w1, const float* __restrict__ b1,
                            const unsigned short* __restrict__ w1b,
                            const unsigned short* __restrict__ w2b, const float* __restrict__ b2,
                            const float* __restrict__ g1, const float* __restrict__ be1,
                            const unsigned short* __restrict__ w3b, const float* __restrict__ b3,
                            const unsigned short* __restrict__ w4b, const float* __restrict__ b4,
                            const float* __restrict__ g2, const float* __restrict__ be2,
                            float* __restrict__ adj_out) {
  const int jc = blockIdx.x, i = blockIdx.y, bz = blockIdx.z;
  const int j0 = jc * 64;
  const int tid = threadIdx.x;
  const int w = tid >> 5, lane = tid & 31, ln = lane & 15, hi = lane >> 4;

  __shared__ __align__(16) unsigned short shA[64 * 64];      // adj_ij bf16
  __shared__ __align__(16) unsigned short shB[64 * 64];      // adj_ji bf16
  __shared__ __align__(16) unsigned short shX[64 * 128];     // x_j bf16
  __shared__ __align__(16) unsigned short shH[4 * 16 * 128]; // per-wave scratch
  __shared__ float shXi[64];

  // ---- stage adj_ij (contiguous 64x64, fp32 -> bf16)
  {
    const GAS f4_t* src = g4(adj + (((size_t)(bz * NN + i)) * NN + j0) * ED);
    unsigned* dst = (unsigned*)shA;
#pragma unroll
    for (int q = 0; q < 8; ++q) {
      const int idx = tid + q * 128;
      f4_t f = src[idx];
      dst[idx * 2 + 0] = pack2(f[0], f[1]);
      dst[idx * 2 + 1] = pack2(f[2], f[3]);
    }
  }
  // ---- stage adj_ji (transpose gather, 64 rows of 256B)
  {
    unsigned* dst = (unsigned*)shB;
#pragma unroll
    for (int q = 0; q < 8; ++q) {
      const int idx = tid + q * 128;
      const int row = idx >> 4, c4 = idx & 15;
      const GAS f4_t* src =
          g4(adj + (((size_t)(bz * NN + j0 + row)) * NN + i) * ED);
      f4_t f = src[c4];
      dst[idx * 2 + 0] = pack2(f[0], f[1]);
      dst[idx * 2 + 1] = pack2(f[2], f[3]);
    }
  }
  // ---- stage x_j (contiguous 64x128, already bf16: straight copy)
  {
    const GAS us8_t* src = gu8(xfb + ((size_t)(bz * NN + j0)) * ND);
    us8_t* dst = (us8_t*)shX;
#pragma unroll
    for (int q = 0; q < 8; ++q)
      dst[tid + q * 128] = src[tid + q * 128];
  }
  // ---- x_i contribution folded with bias
  if (tid < 64) {
    const GAS float* xi = gp(xf + ((size_t)(bz * NN + i)) * ND);
    const GAS float* wr = gp(w1 + tid * EH1 + 256);
    float acc = gload(b1 + tid);
    for (int k = 0; k < ND; ++k) acc += xi[k] * wr[k];
    shXi[tid] = acc;
  }
  __syncthreads();

  const int m0 = w * 16;
  const unsigned short* aRow = shA + (m0 + ln) * 64;
  const unsigned short* bRow = shB + (m0 + ln) * 64;
  const unsigned short* xRow = shX + (m0 + ln) * 128;
  unsigned short* hBase = shH + w * 16 * 128;
  const unsigned short* hRow = hBase + ln * 128;

  // ================= GEMM1: h1 = relu(ua @ W1^T + b1) ======================
  v8f c1[4];
#pragma unroll
  for (int nt = 0; nt < 4; ++nt) {
    const float init = shXi[nt * 16 + ln];
#pragma unroll
    for (int r = 0; r < 8; ++r) c1[nt][r] = init;
  }
#pragma unroll
  for (int kc = 0; kc < 2; ++kc) {   // adj_ij block (K 0..63)
    v16bf a = afrag_lds(aRow + kc * 32, hi);
#pragma unroll
    for (int nt = 0; nt < 4; ++nt)
      c1[nt] = wmma_bf16(a, bfrag_b(w1b + (nt * 16 + ln) * EH1 + kc * 32 + hi * 16), c1[nt]);
  }
#pragma unroll
  for (int kc = 0; kc < 2; ++kc) {   // adj_ji block (K 64..127)
    v16bf a = afrag_lds(bRow + kc * 32, hi);
#pragma unroll
    for (int nt = 0; nt < 4; ++nt)
      c1[nt] = wmma_bf16(a, bfrag_b(w1b + (nt * 16 + ln) * EH1 + 64 + kc * 32 + hi * 16), c1[nt]);
  }
#pragma unroll
  for (int kc = 0; kc < 4; ++kc) {   // x_j block (K 128..255)
    v16bf a = afrag_lds(xRow + kc * 32, hi);
#pragma unroll
    for (int nt = 0; nt < 4; ++nt)
      c1[nt] = wmma_bf16(a, bfrag_b(w1b + (nt * 16 + ln) * EH1 + 128 + kc * 32 + hi * 16), c1[nt]);
  }
  // relu -> LDS bf16
#pragma unroll
  for (int nt = 0; nt < 4; ++nt)
#pragma unroll
    for (int r = 0; r < 8; ++r)
      hBase[(r + 8 * hi) * 128 + nt * 16 + ln] = bfbits(fmaxf(c1[nt][r], 0.f));
  __syncthreads();

  // ================= GEMM2: h2 = h1 @ W2^T + b2 + adj_ij; LN1 ==============
  v8f c2[4];
#pragma unroll
  for (int nt = 0; nt < 4; ++nt) {
    const float init = gload(b2 + nt * 16 + ln);
#pragma unroll
    for (int r = 0; r < 8; ++r) c2[nt][r] = init;
  }
#pragma unroll
  for (int kc = 0; kc < 2; ++kc) {
    v16bf a = afrag_lds(hRow + kc * 32, hi);
#pragma unroll
    for (int nt = 0; nt < 4; ++nt)
      c2[nt] = wmma_bf16(a, bfrag_b(w2b + (nt * 16 + ln) * ED + kc * 32 + hi * 16), c2[nt]);
  }
  // residual with original fp32 adj
#pragma unroll
  for (int nt = 0; nt < 4; ++nt)
#pragma unroll
    for (int r = 0; r < 8; ++r)
      c2[nt][r] += gload(adj + (((size_t)(bz * NN + i)) * NN + (j0 + m0 + r + 8 * hi)) * ED +
                         nt * 16 + ln);
  // LayerNorm over 64 channels (4 N-tiles x 16 lanes per row)
  float mean[8], rstd[8];
#pragma unroll
  for (int r = 0; r < 8; ++r) {
    float s  = c2[0][r] + c2[1][r] + c2[2][r] + c2[3][r];
    float s2 = c2[0][r] * c2[0][r] + c2[1][r] * c2[1][r] +
               c2[2][r] * c2[2][r] + c2[3][r] * c2[3][r];
    s  += __shfl_xor(s, 1, 32);  s  += __shfl_xor(s, 2, 32);
    s  += __shfl_xor(s, 4, 32);  s  += __shfl_xor(s, 8, 32);
    s2 += __shfl_xor(s2, 1, 32); s2 += __shfl_xor(s2, 2, 32);
    s2 += __shfl_xor(s2, 4, 32); s2 += __shfl_xor(s2, 8, 32);
    const float mu = s * (1.f / 64.f);
    const float va = s2 * (1.f / 64.f) - mu * mu;
    mean[r] = mu;
    rstd[r] = rsqrtf(va + EPS);
  }
  float aln[4][8];
#pragma unroll
  for (int nt = 0; nt < 4; ++nt) {
    const float gg = gload(g1 + nt * 16 + ln), bb = gload(be1 + nt * 16 + ln);
#pragma unroll
    for (int r = 0; r < 8; ++r)
      aln[nt][r] = (c2[nt][r] - mean[r]) * rstd[r] * gg + bb;
  }
  __syncthreads();
#pragma unroll
  for (int nt = 0; nt < 4; ++nt)
#pragma unroll
    for (int r = 0; r < 8; ++r)
      hBase[(r + 8 * hi) * 128 + nt * 16 + ln] = bfbits(aln[nt][r]);
  __syncthreads();

  // ================= GEMM3: h3 = relu(aln @ W3^T + b3) (64->128) ===========
  v8f c3[8];
#pragma unroll
  for (int nt = 0; nt < 8; ++nt) {
    const float init = gload(b3 + nt * 16 + ln);
#pragma unroll
    for (int r = 0; r < 8; ++r) c3[nt][r] = init;
  }
#pragma unroll
  for (int kc = 0; kc < 2; ++kc) {
    v16bf a = afrag_lds(hRow + kc * 32, hi);
#pragma unroll
    for (int nt = 0; nt < 8; ++nt)
      c3[nt] = wmma_bf16(a, bfrag_b(w3b + (nt * 16 + ln) * ED + kc * 32 + hi * 16), c3[nt]);
  }
  __syncthreads();
#pragma unroll
  for (int nt = 0; nt < 8; ++nt)
#pragma unroll
    for (int r = 0; r < 8; ++r)
      hBase[(r + 8 * hi) * 128 + nt * 16 + ln] = bfbits(fmaxf(c3[nt][r], 0.f));
  __syncthreads();

  // ================= GEMM4: out = LN2(aln + h3 @ W4^T + b4) ================
  v8f c4[4];
#pragma unroll
  for (int nt = 0; nt < 4; ++nt) {
    const float init = gload(b4 + nt * 16 + ln);
#pragma unroll
    for (int r = 0; r < 8; ++r) c4[nt][r] = init;
  }
#pragma unroll
  for (int kc = 0; kc < 4; ++kc) {
    v16bf a = afrag_lds(hRow + kc * 32, hi);
#pragma unroll
    for (int nt = 0; nt < 4; ++nt)
      c4[nt] = wmma_bf16(a, bfrag_b(w4b + (nt * 16 + ln) * EH2 + kc * 32 + hi * 16), c4[nt]);
  }
#pragma unroll
  for (int nt = 0; nt < 4; ++nt)
#pragma unroll
    for (int r = 0; r < 8; ++r) c4[nt][r] += aln[nt][r];
#pragma unroll
  for (int r = 0; r < 8; ++r) {
    float s  = c4[0][r] + c4[1][r] + c4[2][r] + c4[3][r];
    float s2 = c4[0][r] * c4[0][r] + c4[1][r] * c4[1][r] +
               c4[2][r] * c4[2][r] + c4[3][r] * c4[3][r];
    s  += __shfl_xor(s, 1, 32);  s  += __shfl_xor(s, 2, 32);
    s  += __shfl_xor(s, 4, 32);  s  += __shfl_xor(s, 8, 32);
    s2 += __shfl_xor(s2, 1, 32); s2 += __shfl_xor(s2, 2, 32);
    s2 += __shfl_xor(s2, 4, 32); s2 += __shfl_xor(s2, 8, 32);
    const float mu = s * (1.f / 64.f);
    const float va = s2 * (1.f / 64.f) - mu * mu;
    mean[r] = mu;
    rstd[r] = rsqrtf(va + EPS);
  }
#pragma unroll
  for (int nt = 0; nt < 4; ++nt) {
    const float gg = gload(g2 + nt * 16 + ln), bb = gload(be2 + nt * 16 + ln);
#pragma unroll
    for (int r = 0; r < 8; ++r) {
      const size_t off = (((size_t)(bz * NN + i)) * NN + (j0 + m0 + r + 8 * hi)) * ED +
                         nt * 16 + ln;
      gpw(adj_out)[off] = (c4[nt][r] - mean[r]) * rstd[r] * gg + bb;
    }
  }
}

// ---------------------------------------------------------------------------
extern "C" void kernel_launch(void* const* d_in, const int* in_sizes, int n_in,
                              void* d_out, int out_size, void* d_ws, size_t ws_size,
                              hipStream_t stream) {
  (void)in_sizes; (void)n_in; (void)out_size; (void)ws_size;
  const float* x      = (const float*)d_in[0];
  const float* adj    = (const float*)d_in[1];
  const float* wnq    = (const float*)d_in[2];
  const float* wnk    = (const float*)d_in[3];
  const float* wnv    = (const float*)d_in[4];
  const float* weq    = (const float*)d_in[5];
  const float* wek    = (const float*)d_in[6];
  const float* wev    = (const float*)d_in[7];
  const float* nfc1_w = (const float*)d_in[8];
  const float* nfc1_b = (const float*)d_in[9];
  const float* ln1_g  = (const float*)d_in[10];
  const float* ln1_b  = (const float*)d_in[11];
  const float* nfc2_w = (const float*)d_in[12];
  const float* nfc2_b = (const float*)d_in[13];
  const float* nfc3_w = (const float*)d_in[14];
  const float* nfc3_b = (const float*)d_in[15];
  const float* ln2_g  = (const float*)d_in[16];
  const float* ln2_b  = (const float*)d_in[17];
  const float* efc1_w = (const float*)d_in[18];
  const float* efc1_b = (const float*)d_in[19];
  const float* efc2_w = (const float*)d_in[20];
  const float* efc2_b = (const float*)d_in[21];
  const float* eln1_g = (const float*)d_in[22];
  const float* eln1_b = (const float*)d_in[23];
  const float* efc3_w = (const float*)d_in[24];
  const float* efc3_b = (const float*)d_in[25];
  const float* efc4_w = (const float*)d_in[26];
  const float* efc4_b = (const float*)d_in[27];
  const float* eln2_g = (const float*)d_in[28];
  const float* eln2_b = (const float*)d_in[29];

  float* ws = (float*)d_ws;
  float* qn = ws;
  float* kn = ws + (size_t)BB * NN * ND;
  float* vn = ws + 2 * (size_t)BB * NN * ND;
  float* ao = ws + 3 * (size_t)BB * NN * ND;
  unsigned short* wbf = (unsigned short*)(ws + 4 * (size_t)BB * NN * ND);
  // bf16 weight offsets (ushorts)
  unsigned short* wqb  = wbf + 0;
  unsigned short* wkb  = wbf + 16384;
  unsigned short* wvb  = wbf + 32768;
  unsigned short* weqb = wbf + 49152;
  unsigned short* wekb = wbf + 57344;
  unsigned short* wevb = wbf + 65536;
  unsigned short* w1b  = wbf + 73728;   // efc1 64x384
  unsigned short* w2b  = wbf + 98304;   // efc2 64x64
  unsigned short* w3b  = wbf + 102400;  // efc3 128x64
  unsigned short* w4b  = wbf + 110592;  // efc4 64x128
  unsigned short* xfb  = wbf + 118784;  // bf16 copy of x_out (768x128)

  float* xout   = (float*)d_out;
  float* adjout = xout + (size_t)BB * NN * ND;

  CvtDesc desc;
  const float* srcs[10] = { wnq, wnk, wnv, weq, wek, wev,
                            efc1_w, efc2_w, efc3_w, efc4_w };
  const unsigned cnt4[10] = { 4096, 4096, 4096, 2048, 2048, 2048,
                              6144, 1024, 2048, 2048 };
  const unsigned dsh[10]  = { 0, 16384, 32768, 49152, 57344, 65536,
                              73728, 98304, 102400, 110592 };
  for (int s = 0; s < 10; ++s) {
    desc.src[s]  = srcs[s];
    desc.cnt4[s] = cnt4[s];
    desc.dsh[s]  = dsh[s];
  }

  cvt_weights_kernel<<<dim3(64), dim3(256), 0, stream>>>(desc, wbf);

  node_proj_kernel<<<dim3(BB * NN / 16), dim3(32), 0, stream>>>(
      x, wqb, wkb, wvb, qn, kn, vn);

  attention_kernel<<<dim3(NN, BB), dim3(256), 0, stream>>>(
      adj, qn, kn, vn, weqb, wekb, wevb, ao);

  node_post_kernel<<<dim3(BB * NN), dim3(256), 0, stream>>>(
      x, ao, nfc1_w, nfc1_b, ln1_g, ln1_b, nfc2_w, nfc2_b, nfc3_w, nfc3_b,
      ln2_g, ln2_b, xout, xfb);

  edge_kernel<<<dim3(NN / 64, NN, BB), dim3(128), 0, stream>>>(
      adj, xout, xfb, efc1_w, efc1_b, w1b, w2b, efc2_b, eln1_g, eln1_b,
      w3b, efc3_b, w4b, efc4_b, eln2_g, eln2_b, adjout);
}
